// SelfAttention_8710193676497
// MI455X (gfx1250) — compile-verified, matrix operand order checked by reading, not run on hardware
//
#include <hip/hip_runtime.h>
#include <hip/hip_bf16.h>

// ---------------------------------------------------------------------------
// Problem constants (match reference)
// ---------------------------------------------------------------------------
constexpr int B_  = 16;
constexpr int C_  = 256;
constexpr int L_  = 2048;
constexpr int DK_ = 32;
constexpr int MW_ = 320;   // rows of concatenated [Wk; Wq; Wv]

// ---------------------------------------------------------------------------
// CDNA5 WMMA types
// ---------------------------------------------------------------------------
typedef __bf16 bf16_t;
typedef bf16_t v16bf __attribute__((ext_vector_type(16)));
typedef float  v8f   __attribute__((ext_vector_type(8)));

union Frag {
    v16bf          v;
    unsigned int   w[8];
    unsigned short u[16];
};

__device__ __forceinline__ v8f wmma_bf16(const Frag& a, const Frag& b, v8f c) {
    // D = A(16x32 bf16) * B(32x16 bf16) + C(16x16 f32)
    return __builtin_amdgcn_wmma_f32_16x16x32_bf16(
        /*neg_a=*/false, a.v, /*neg_b=*/false, b.v,
        /*c_mod=*/(short)0, c, /*reuse_a=*/false, /*reuse_b=*/false);
}

// f32 -> bf16, round-to-nearest-even (only used off the hot GEMM loops)
__device__ __forceinline__ unsigned short f2bf(float f) {
    unsigned u = __float_as_uint(f);
    u += 0x7FFFu + ((u >> 16) & 1u);
    return (unsigned short)(u >> 16);
}

// A/B fragment 16-bit pair index map (ISA 7.12.2, 16-bit A-matrix 16x32):
// lanes 0-15: VGPR j<4 -> K pair base 2j       ; j>=4 -> 16 + 2(j-4)
// lanes16-31: VGPR j<4 -> K pair base 8 + 2j   ; j>=4 -> 24 + 2(j-4)
// Returned in u32 (bf16-pair) units within a 16-u32 (32 bf16) row.
__device__ __forceinline__ int kpair_idx(int j, int hf) {
    return (j < 4) ? (4 * hf + j) : (8 + 4 * hf + (j - 4));
}

// Cross-lane reductions within 16-lane halves (xor masks <= 8 stay in half).
#define SWZ_XOR(x, m) \
    __int_as_float(__builtin_amdgcn_ds_swizzle(__float_as_int(x), (((m) << 10) | 0x1F)))

__device__ __forceinline__ float red_max16(float x) {
    x = fmaxf(x, SWZ_XOR(x, 1));
    x = fmaxf(x, SWZ_XOR(x, 2));
    x = fmaxf(x, SWZ_XOR(x, 4));
    x = fmaxf(x, SWZ_XOR(x, 8));
    return x;
}
__device__ __forceinline__ float red_sum16(float x) {
    x += SWZ_XOR(x, 1);
    x += SWZ_XOR(x, 2);
    x += SWZ_XOR(x, 4);
    x += SWZ_XOR(x, 8);
    return x;
}

// ---------------------------------------------------------------------------
// Kernel 0: one-shot pack of [Wk; Wq; Wv] (320x256 f32) -> bf16 Wc.
// 160 KB result stays resident in the 192 MB L2 for the projection GEMMs.
// ---------------------------------------------------------------------------
__global__ void __launch_bounds__(256)
pack_w_kernel(const float* __restrict__ Wq, const float* __restrict__ Wk,
              const float* __restrict__ Wv, unsigned short* __restrict__ Wc)
{
    const int i = blockIdx.x * 256 + threadIdx.x;   // 0 .. 320*256-1
    const int r = i >> 8, c = i & 255;
    float v;
    if (r < 32)      v = Wk[r * C_ + c];            // rows  0..31  -> queries (g)
    else if (r < 64) v = Wq[(r - 32) * C_ + c];     // rows 32..63  -> keys    (f)
    else             v = Wv[(size_t)(r - 64) * C_ + c];  // rows 64..319 -> values (h)
    Wc[i] = f2bf(v);
}

// ---------------------------------------------------------------------------
// Kernel 1: projections.  Y = Wc (320x256 bf16) * x_b (256x2048),
// stored transposed in bf16:  qT[b][l][0:32]  = (Wk x)^T   (queries = g)
//                             kT[b][l][0:32]  = (Wq x)^T   (keys    = f)
//                             vT[b][l][0:256] = (Wv x)^T   (values  = h)
// Grid: (L/64, B), 128 threads (4 waves). Each wave owns 5 of 20 M-tiles.
// ---------------------------------------------------------------------------
__global__ void __launch_bounds__(128)
proj_kernel(const float* __restrict__ x,
            const unsigned short* __restrict__ Wc,
            unsigned short* __restrict__ qT, unsigned short* __restrict__ kT,
            unsigned short* __restrict__ vT)
{
    __shared__ unsigned short lds_x[64][256];  // [n_local][c] bf16 (32 KB)

    const int tid  = threadIdx.x;
    const int wave = tid >> 5;
    const int lane = tid & 31;
    const int hf   = lane >> 4;
    const int lr   = lane & 15;
    const int b      = blockIdx.y;
    const int nblock = blockIdx.x * 64;

    // Stage x block (256 x 64 f32 -> bf16, transposed to [n][c]); coalesced rows.
    for (int idx = tid; idx < 256 * 64; idx += 128) {
        const int c = idx >> 6, n = idx & 63;
        lds_x[n][c] = f2bf(x[((size_t)b * C_ + c) * L_ + nblock + n]);
    }
    __syncthreads();
    const unsigned* lds_x32 = (const unsigned*)&lds_x[0][0];  // row = 128 u32
    const unsigned* Wc32    = (const unsigned*)Wc;            // row = 128 u32

    for (int mt = wave; mt < MW_ / 16; mt += 4) {
        v8f acc0 = {}, acc1 = {}, acc2 = {}, acc3 = {};
        const int arow = mt * 16 + lr;  // A-fragment: lane = M (Wc row)

        #pragma unroll
        for (int ks = 0; ks < 8; ++ks) {           // K = 256 in steps of 32
            Frag A;                                 // straight bf16-pair loads
            #pragma unroll
            for (int j = 0; j < 8; ++j)
                A.w[j] = Wc32[(size_t)arow * 128 + ks * 16 + kpair_idx(j, hf)];
            #pragma unroll
            for (int nt = 0; nt < 4; ++nt) {
                Frag Bf;                            // B-fragment: lane = N (x col)
                const int nrow = nt * 16 + lr;
                #pragma unroll
                for (int j = 0; j < 8; ++j)
                    Bf.w[j] = lds_x32[nrow * 128 + ks * 16 + kpair_idx(j, hf)];
                v8f& acc = (nt == 0) ? acc0 : (nt == 1) ? acc1 : (nt == 2) ? acc2 : acc3;
                acc = wmma_bf16(A, Bf, acc);
            }
        }

        // Store D tiles (lane = N, VGPR r -> M = r + 8*half) transposed as bf16.
        #pragma unroll
        for (int nt = 0; nt < 4; ++nt) {
            const v8f& acc = (nt == 0) ? acc0 : (nt == 1) ? acc1 : (nt == 2) ? acc2 : acc3;
            const int n = nblock + nt * 16 + lr;
            #pragma unroll
            for (int v = 0; v < 8; ++v) {
                const int r = mt * 16 + v + 8 * hf;        // row in [Wk;Wq;Wv]
                const unsigned short val = f2bf(acc[v]);
                if (r < 32)
                    qT[((size_t)b * L_ + n) * DK_ + r] = val;
                else if (r < 64)
                    kT[((size_t)b * L_ + n) * DK_ + (r - 32)] = val;
                else
                    vT[((size_t)b * L_ + n) * C_ + (r - 64)] = val;
            }
        }
    }
}

// ---------------------------------------------------------------------------
// Kernel 2: flash attention over keys l for a 16-query block m.
// Grid: (L/16, B), 128 threads (4 waves).
//   wave 0 : S = Q K^T (d=32 -> one WMMA per 16x16 tile), online softmax,
//            publishes bf16 P(16x32) + per-row rescale alpha via LDS.
//   waves 0-3 : each owns 4 c-tiles (64 channels) of O = P V.
// Epilogue goes through LDS so the 67 MB out-store / x-load stream is
// emitted as contiguous 64B segments instead of per-lane scatters.
// ---------------------------------------------------------------------------
__global__ void __launch_bounds__(128)
attn_kernel(const unsigned short* __restrict__ qT,
            const unsigned short* __restrict__ kT,
            const unsigned short* __restrict__ vT,
            const float* __restrict__ x,
            const float* __restrict__ gamma,
            float* __restrict__ out)
{
    __shared__ unsigned short lds_v[256][32];  // V chunk transposed [c][l_local], 16 KB
    __shared__ unsigned short lds_p[16][32];   // P tile [m][l_local], bf16
    __shared__ float          lds_a[16];       // per-row rescale / inv-sigma
    __shared__ float          lds_o[16][260];  // epilogue tile [m][c], padded rows

    const int tid  = threadIdx.x;
    const int wave = tid >> 5;
    const int lane = tid & 31;
    const int hf   = lane >> 4;
    const int lr   = lane & 15;
    const int b      = blockIdx.y;
    const int mblock = blockIdx.x * 16;

    const unsigned* qT32 = (const unsigned*)qT;  // rows of 16 u32 (32 bf16)
    const unsigned* kT32 = (const unsigned*)kT;
    const unsigned* vT32 = (const unsigned*)vT;  // rows of 128 u32 (256 bf16)

    // Q fragment (only wave 0 uses it): lane = query m, K = d in [0,32)
    Frag Qf;
    if (wave == 0) {
        const size_t row = (size_t)b * L_ + mblock + lr;
        #pragma unroll
        for (int j = 0; j < 8; ++j) Qf.w[j] = qT32[row * 16 + kpair_idx(j, hf)];
    }

    float mu[8], sg[8];
    #pragma unroll
    for (int r = 0; r < 8; ++r) { mu[r] = -1e30f; sg[r] = 0.0f; }
    v8f acc[4] = {v8f{}, v8f{}, v8f{}, v8f{}};

    constexpr int NKB = L_ / 32;
    for (int kb = 0; kb < NKB; ++kb) {
        const int lbase = kb * 32;
        __syncthreads();  // previous chunk's LDS consumers done

        // Stage V chunk 32 x 256 bf16, transposed into lds_v[c][ll] (coalesced loads).
        for (int idx = tid; idx < 32 * 128; idx += 128) {
            const int ll = idx >> 7, t = idx & 127;
            const unsigned val = vT32[((size_t)b * L_ + lbase + ll) * 128 + t];
            lds_v[2 * t][ll]     = (unsigned short)val;
            lds_v[2 * t + 1][ll] = (unsigned short)(val >> 16);
        }
        if (kb + 1 < NKB)  // pull next V chunk toward L2 (global_prefetch_b8)
            __builtin_prefetch(&vT32[((size_t)b * L_ + lbase + 32 + (tid >> 2)) * 128 +
                                     (tid & 3) * 32], 0, 0);
        __syncthreads();

        if (wave == 0) {
            // Score tiles: D[m][l] = sum_d Q[m][d] * kT[l][d]
            Frag K0, K1;
            const size_t r0 = (size_t)b * L_ + lbase + lr;
            #pragma unroll
            for (int j = 0; j < 8; ++j) {
                const int kp = kpair_idx(j, hf);
                K0.w[j] = kT32[r0 * 16 + kp];
                K1.w[j] = kT32[(r0 + 16) * 16 + kp];
            }
            v8f s0 = wmma_bf16(Qf, K0, v8f{});
            v8f s1 = wmma_bf16(Qf, K1, v8f{});

            // Online softmax over keys (N dim = across 16 lanes of each half).
            #pragma unroll
            for (int r = 0; r < 8; ++r) {
                float cm = red_max16(fmaxf(s0[r], s1[r]));
                const float nm = fmaxf(mu[r], cm);
                const float al = __expf(mu[r] - nm);
                const float p0 = __expf(s0[r] - nm);
                const float p1 = __expf(s1[r] - nm);
                const float rs = red_sum16(p0 + p1);
                sg[r] = sg[r] * al + rs;
                mu[r] = nm;
                const int m = r + 8 * hf;
                lds_p[m][lr]      = f2bf(p0);
                lds_p[m][16 + lr] = f2bf(p1);
                lds_a[m] = al;  // 16 lanes write identical value
            }
        }
        __syncthreads();

        // All waves: O_tile = alpha * O_tile + P * Vchunk for 4 c-tiles each.
        float al[8];
        #pragma unroll
        for (int r = 0; r < 8; ++r) al[r] = lds_a[r + 8 * hf];

        Frag Pf;  // A-fragment of P: lane = m, K = l_local
        const unsigned* lds_p32 = (const unsigned*)&lds_p[0][0];  // rows of 16 u32
        #pragma unroll
        for (int j = 0; j < 8; ++j) Pf.w[j] = lds_p32[lr * 16 + kpair_idx(j, hf)];

        const unsigned* lds_v32 = (const unsigned*)&lds_v[0][0];  // rows of 16 u32
        #pragma unroll
        for (int jj = 0; jj < 4; ++jj) {
            const int c = (wave * 4 + jj) * 16 + lr;  // B-fragment: lane = N = channel
            Frag Vf;
            #pragma unroll
            for (int j = 0; j < 8; ++j) Vf.w[j] = lds_v32[c * 16 + kpair_idx(j, hf)];
            #pragma unroll
            for (int r = 0; r < 8; ++r) acc[jj][r] *= al[r];
            acc[jj] = wmma_bf16(Pf, Vf, acc[jj]);
        }
    }

    __syncthreads();
    if (wave == 0) {
        #pragma unroll
        for (int r = 0; r < 8; ++r) lds_a[r + 8 * hf] = 1.0f / sg[r];
    }
    __syncthreads();

    // Stage gamma * O[m][c] / sigma[m] into LDS (D-layout -> [m][c] tile).
    const float g = gamma[0];
    #pragma unroll
    for (int jj = 0; jj < 4; ++jj) {
        const int c = (wave * 4 + jj) * 16 + lr;
        #pragma unroll
        for (int r = 0; r < 8; ++r) {
            const float inv = lds_a[r + 8 * hf];
            lds_o[r + 8 * hf][c] = g * acc[jj][r] * inv;
        }
    }
    __syncthreads();

    // out[b][c][mblock+m] = lds_o[m][c] + x[b][c][mblock+m]; 16 consecutive
    // threads walk 16 consecutive m -> 64B contiguous load+store segments.
    for (int idx = tid; idx < 16 * 256; idx += 128) {
        const int m = idx & 15, c = idx >> 4;
        const size_t o = ((size_t)b * C_ + c) * L_ + mblock + m;
        out[o] = lds_o[m][c] + x[o];
    }
}

// ---------------------------------------------------------------------------
// Launch. Workspace (bf16): qT (2 MB) | kT (2 MB) | vT (16 MB) | Wc (160 KB).
// ---------------------------------------------------------------------------
extern "C" void kernel_launch(void* const* d_in, const int* in_sizes, int n_in,
                              void* d_out, int out_size, void* d_ws, size_t ws_size,
                              hipStream_t stream) {
    (void)in_sizes; (void)n_in; (void)out_size; (void)ws_size;
    const float* x     = (const float*)d_in[0];
    const float* Wq    = (const float*)d_in[1];
    const float* Wk    = (const float*)d_in[2];
    const float* Wv    = (const float*)d_in[3];
    const float* gamma = (const float*)d_in[4];
    float* out = (float*)d_out;

    unsigned short* qT = (unsigned short*)d_ws;
    unsigned short* kT = qT + (size_t)B_ * L_ * DK_;
    unsigned short* vT = kT + (size_t)B_ * L_ * DK_;
    unsigned short* Wc = vT + (size_t)B_ * L_ * C_;

    pack_w_kernel<<<dim3(MW_ * C_ / 256), 256, 0, stream>>>(Wq, Wk, Wv, Wc);
    proj_kernel<<<dim3(L_ / 64, B_), 128, 0, stream>>>(x, Wc, qT, kT, vT);
    attn_kernel<<<dim3(L_ / 16, B_), 128, 0, stream>>>(qT, kT, vT, x, gamma, out);
}
